// LLaMAAttention_19885698580836
// MI455X (gfx1250) — matrix operand order repaired and checked by reading, not compiled, not run on hardware
//
#include <hip/hip_runtime.h>
#include <hip/hip_bf16.h>

typedef __attribute__((ext_vector_type(16))) __bf16 v16bf;
typedef __attribute__((ext_vector_type(8)))  float  v8f;
typedef __attribute__((ext_vector_type(4)))  unsigned int u32x4;
typedef __attribute__((ext_vector_type(8)))  unsigned int u32x8;

constexpr int S_LEN = 2048;
constexpr int HID   = 4096;
constexpr int NH    = 32;
constexpr int NKV   = 8;
constexpr int HD    = 128;

__device__ inline v8f wmma_bf16(v16bf a, v16bf b, v8f c) {
  return __builtin_amdgcn_wmma_f32_16x16x32_bf16(false, a, false, b, (short)0, c, false, false);
}

// A-fragment (16x32): lane m = lane&15, half = lane>>4,
// pair vg -> K0 = (vg/4)*16 + half*8 + (vg%4)*2   (ISA 7.12.2, 16-bit A)
__device__ inline v16bf load_a_frag(const __bf16* base, int stride, int lane) {
  int m = lane & 15, half = lane >> 4;
  v16bf a;
#pragma unroll
  for (int vg = 0; vg < 8; ++vg) {
    int k0 = ((vg >> 2) << 4) + (half << 3) + ((vg & 3) << 1);
    a[2 * vg]     = base[m * stride + k0];
    a[2 * vg + 1] = base[m * stride + k0 + 1];
  }
  return a;
}

// B-fragment (32x16): lane n = lane&15, half = lane>>4, K0 = (vg/4)*16 + half*8 + (vg%4)*2.
// base laid out [n][k] (transposed operand).
__device__ inline v16bf load_b_frag(const __bf16* base, int stride, int lane) {
  int n = lane & 15, half = lane >> 4;
  v16bf b;
#pragma unroll
  for (int vg = 0; vg < 8; ++vg) {
    int k0 = ((vg >> 2) << 4) + (half << 3) + ((vg & 3) << 1);
    b[2 * vg]     = base[n * stride + k0];
    b[2 * vg + 1] = base[n * stride + k0 + 1];
  }
  return b;
}

// Low 32 bits of a generic pointer into LDS = LDS byte offset (ISA 10.2).
__device__ inline unsigned lds_off_u32(const void* p) {
  return (unsigned)(unsigned long long)p;
}

// ---------------------------------------------------------------------------
// TDM: DMA one [rows x cols] bf16 2D tile from global to LDS.
// pad_code: LDS pad of 1 DWORD every (2<<pad_code) DWORDs.
//   cols=32  -> pad_code 3 (16 dw = one row)  -> LDS row stride 34 elems
//   cols=128 -> pad_code 5 (64 dw = one row)  -> LDS row stride 130 elems
// ---------------------------------------------------------------------------
__device__ inline void tdm_load_tile_bf16(unsigned lds_byte_off,
                                          const __bf16* tile_ptr,
                                          unsigned rows, unsigned cols,
                                          unsigned row_stride,
                                          unsigned tensor_d0, unsigned tensor_d1,
                                          unsigned pad_code) {
  unsigned long long ga = (unsigned long long)tile_ptr;
  u32x4 g0;
  g0[0] = 1u;                                                // count=1 (user D#)
  g0[1] = lds_byte_off;                                      // lds_addr
  g0[2] = (unsigned)(ga & 0xffffffffu);                      // global_addr[31:0]
  g0[3] = (unsigned)((ga >> 32) & 0x01ffffffu) | (2u << 30); // ga[56:32] | type=2
  u32x8 g1;
  g1[0] = (1u << 16) | (1u << 20) | (pad_code << 22);        // 2B, pad_en, interval
  g1[1] = (tensor_d0 & 0xffffu) << 16;                       // tensor_dim0[15:0]
  g1[2] = (tensor_d0 >> 16) | ((tensor_d1 & 0xffffu) << 16); // d0[31:16], d1[15:0]
  g1[3] = (tensor_d1 >> 16) | (cols << 16);                  // d1[31:16], tile_dim0
  g1[4] = rows;                                              // tile_dim1 (dim2=0)
  g1[5] = row_stride;                                        // dim0_stride[31:0]
  g1[6] = 0;
  g1[7] = 0;
  u32x4 gz = {0u, 0u, 0u, 0u};
  asm volatile("tensor_load_to_lds %0, %1, %2, %3"
               :: "s"(g0), "s"(g1), "s"(gz), "s"(gz) : "memory");
}

// ---------------------------------------------------------------------------
// bf16 GEMM with TDM staging:  C[M,N] = A[M,Kd] * Bt[N,Kd]^T
// BM=BN=128, BK=32; 128 threads = 4 waves (2x2), each wave computes 64x64.
// ---------------------------------------------------------------------------
#define GBM 128
#define GBN 128
#define GBK 32
#define LSTR 34   // 32 + 1 dword pad

__global__ __launch_bounds__(128) void gemm_bf16_tdm(
    const __bf16* __restrict__ A, const __bf16* __restrict__ Bt,
    float* __restrict__ C, int M, int N, int Kd) {
  __shared__ __bf16 As[2][GBM][LSTR];
  __shared__ __bf16 Bs[2][GBN][LSTR];

  const int t = threadIdx.x;
  const int lane = t & 31, w = t >> 5;
  const int wr = w >> 1, wc = w & 1;
  const int m0 = blockIdx.y * GBM, n0 = blockIdx.x * GBN;
  const int nk = Kd / GBK;

  const v8f vzero = {0.f, 0.f, 0.f, 0.f, 0.f, 0.f, 0.f, 0.f};
  v8f acc[4][4];
#pragma unroll
  for (int i = 0; i < 4; ++i)
#pragma unroll
    for (int j = 0; j < 4; ++j) acc[i][j] = vzero;

  if (w == 0) {
    tdm_load_tile_bf16(lds_off_u32(&As[0][0][0]), A + (size_t)m0 * Kd,
                       GBM, GBK, Kd, Kd, M, 3);
    tdm_load_tile_bf16(lds_off_u32(&Bs[0][0][0]), Bt + (size_t)n0 * Kd,
                       GBN, GBK, Kd, Kd, N, 3);
  }

  for (int kb = 0; kb < nk; ++kb) {
    const int cur = kb & 1;
    if (w == 0) {
      if (kb + 1 < nk) {
        tdm_load_tile_bf16(lds_off_u32(&As[cur ^ 1][0][0]),
                           A + (size_t)m0 * Kd + (kb + 1) * GBK,
                           GBM, GBK, Kd, Kd, M, 3);
        tdm_load_tile_bf16(lds_off_u32(&Bs[cur ^ 1][0][0]),
                           Bt + (size_t)n0 * Kd + (kb + 1) * GBK,
                           GBN, GBK, Kd, Kd, N, 3);
        __builtin_amdgcn_s_wait_tensorcnt(2);  // oldest pair (buf cur) done
      } else {
        __builtin_amdgcn_s_wait_tensorcnt(0);
      }
    }
    __syncthreads();

    v16bf af[4], bfr[4];
#pragma unroll
    for (int i = 0; i < 4; ++i)
      af[i] = load_a_frag(&As[cur][wr * 64 + i * 16][0], LSTR, lane);
#pragma unroll
    for (int j = 0; j < 4; ++j)
      bfr[j] = load_b_frag(&Bs[cur][wc * 64 + j * 16][0], LSTR, lane);
#pragma unroll
    for (int i = 0; i < 4; ++i)
#pragma unroll
      for (int j = 0; j < 4; ++j)
        acc[i][j] = wmma_bf16(af[i], bfr[j], acc[i][j]);

    __syncthreads();
  }

  const int n = lane & 15, half = lane >> 4;
#pragma unroll
  for (int i = 0; i < 4; ++i)
#pragma unroll
    for (int j = 0; j < 4; ++j)
#pragma unroll
      for (int r = 0; r < 8; ++r) {
        int mg = m0 + wr * 64 + i * 16 + r + half * 8;
        int ng = n0 + wc * 64 + j * 16 + n;
        C[(size_t)mg * N + ng] = acc[i][j][r];
      }
}

// ---------------------------------------------------------------------------
// fp32 -> bf16 elementwise (vectorized x4)
// ---------------------------------------------------------------------------
__global__ void f32_to_bf16_kernel(const float* __restrict__ src,
                                   __bf16* __restrict__ dst, int n4) {
  int i = blockIdx.x * blockDim.x + threadIdx.x;
  if (i >= n4) return;
  float4 v = reinterpret_cast<const float4*>(src)[i];
  __bf16* d = dst + (size_t)i * 4;
  d[0] = (__bf16)v.x; d[1] = (__bf16)v.y; d[2] = (__bf16)v.z; d[3] = (__bf16)v.w;
}

// ---------------------------------------------------------------------------
// Transpose fp32 [R][C] -> bf16 [C][R] via LDS tile (coalesced both sides)
// ---------------------------------------------------------------------------
__global__ __launch_bounds__(256) void transpose_to_bf16_kernel(
    const float* __restrict__ src, __bf16* __restrict__ dst, int R, int C) {
  __shared__ float tile[32][33];
  int tx = threadIdx.x, ty = threadIdx.y;   // 32 x 8
  int c0 = blockIdx.x * 32, r0 = blockIdx.y * 32;
#pragma unroll
  for (int i = 0; i < 4; ++i)
    tile[ty + 8 * i][tx] = src[(size_t)(r0 + ty + 8 * i) * C + c0 + tx];
  __syncthreads();
#pragma unroll
  for (int i = 0; i < 4; ++i)
    dst[(size_t)(c0 + ty + 8 * i) * R + r0 + tx] = (__bf16)tile[tx][ty + 8 * i];
}

// ---------------------------------------------------------------------------
// Per-KV-head transpose: V fp32 [S][NKV*HD] -> Vt bf16 [NKV][HD][S]
// ---------------------------------------------------------------------------
__global__ __launch_bounds__(256) void v_transpose_kernel(
    const float* __restrict__ V, __bf16* __restrict__ Vt) {
  __shared__ float tile[32][33];
  int tx = threadIdx.x, ty = threadIdx.y;   // 32 x 8
  int h  = blockIdx.z;
  int s0 = blockIdx.x * 32;
  int d0 = blockIdx.y * 32;
#pragma unroll
  for (int i = 0; i < 4; ++i)
    tile[ty + 8 * i][tx] =
        V[(size_t)(s0 + ty + 8 * i) * (NKV * HD) + h * HD + d0 + tx];
  __syncthreads();
#pragma unroll
  for (int i = 0; i < 4; ++i)
    Vt[((size_t)h * HD + d0 + ty + 8 * i) * S_LEN + s0 + tx] =
        (__bf16)tile[tx][ty + 8 * i];
}

// ---------------------------------------------------------------------------
// RoPE: read fp32 Q/K, rotate in fp32, write bf16 Qbf (pre-scaled 1/sqrt(HD))
// and bf16 Kbf.
// ---------------------------------------------------------------------------
__global__ void rope_bf16_kernel(const float* __restrict__ Qin,
                                 const float* __restrict__ Kin,
                                 __bf16* __restrict__ Qout,
                                 __bf16* __restrict__ Kout) {
  int idx = blockIdx.x * blockDim.x + threadIdx.x;
  const int total = S_LEN * (NH + NKV) * (HD / 2);
  if (idx >= total) return;
  int i = idx & 63;
  int rest = idx >> 6;
  int h = rest % (NH + NKV);
  int pos = rest / (NH + NKV);
  float inv_freq = __expf(-(float)i * (9.210340371976184f / 64.0f)); // theta^(-i/64)
  float sn, cs;
  sincosf((float)pos * inv_freq, &sn, &cs);
  const float scale = 0.08838834764831845f; // 1/sqrt(128), folded into Q
  if (h < NH) {
    const float* p = Qin + (size_t)pos * (NH * HD) + h * HD;
    __bf16* o = Qout + (size_t)pos * (NH * HD) + h * HD;
    float x1 = p[i], x2 = p[i + 64];
    o[i]      = (__bf16)((x1 * cs - x2 * sn) * scale);
    o[i + 64] = (__bf16)((x2 * cs + x1 * sn) * scale);
  } else {
    int hk = h - NH;
    const float* p = Kin + (size_t)pos * (NKV * HD) + hk * HD;
    __bf16* o = Kout + (size_t)pos * (NKV * HD) + hk * HD;
    float x1 = p[i], x2 = p[i + 64];
    o[i]      = (__bf16)(x1 * cs - x2 * sn);
    o[i + 64] = (__bf16)(x2 * cs + x1 * sn);
  }
}

// ---------------------------------------------------------------------------
// Flash attention, causal + GQA, fully bf16-fed via TDM.
// grid = (S/64, NH), 128 threads = 4 waves; each wave: 16 query rows.
// K tiles [32 keys][128 feat] and Vt tiles [128 dim][32 keys] are DMA'd by
// the TDM, double-buffered, tracked with TENSORcnt.
// ---------------------------------------------------------------------------
__global__ __launch_bounds__(128) void attn_kernel(
    const __bf16* __restrict__ Qbf,   // [S][NH*HD], pre-scaled
    const __bf16* __restrict__ Kbf,   // [S][NKV*HD]
    const __bf16* __restrict__ Vt,    // [NKV][HD][S]
    __bf16* __restrict__ ctx) {       // [S][NH*HD]
  __shared__ __bf16 Ks[2][32][HD + 2];    // [key][feat], stride 130 (pad code 5)
  __shared__ __bf16 Vs[2][HD][32 + 2];    // [dim][key],  stride 34  (pad code 3)
  __shared__ __bf16 Ps[4][16][32];        // per-wave P tile

  const int t = threadIdx.x, lane = t & 31, w = t >> 5;
  const int head = blockIdx.y, kvh = head >> 2;
  const int qtile = blockIdx.x;
  const int q0 = qtile * 64 + w * 16;
  const int lm = lane & 15, half = lane >> 4;

  // Q fragments straight from bf16 global (A-layout gather)
  v16bf qf[4];
#pragma unroll
  for (int c = 0; c < 4; ++c) {
#pragma unroll
    for (int vg = 0; vg < 8; ++vg) {
      int k0 = c * 32 + ((vg >> 2) << 4) + (half << 3) + ((vg & 3) << 1);
      const __bf16* qp = Qbf + (size_t)(q0 + lm) * (NH * HD) + head * HD + k0;
      qf[c][2 * vg]     = qp[0];
      qf[c][2 * vg + 1] = qp[1];
    }
  }

  const v8f vzero = {0.f, 0.f, 0.f, 0.f, 0.f, 0.f, 0.f, 0.f};
  v8f acc[8];
  float m_i[8], l_i[8];
#pragma unroll
  for (int r = 0; r < 8; ++r) { acc[r] = vzero; m_i[r] = -1e30f; l_i[r] = 0.f; }

  const int nb = (qtile * 64 + 64) >> 5;   // key blocks of 32, uniform in WG
  if (w == 0) {
    tdm_load_tile_bf16(lds_off_u32(&Ks[0][0][0]),
                       Kbf + (size_t)kvh * HD, 32, HD, NKV * HD,
                       NKV * HD, S_LEN, 5);
    tdm_load_tile_bf16(lds_off_u32(&Vs[0][0][0]),
                       Vt + (size_t)kvh * HD * S_LEN, HD, 32, S_LEN,
                       S_LEN, HD, 3);
  }

  for (int b = 0; b < nb; ++b) {
    const int j0 = b << 5;
    const int cur = b & 1;
    if (w == 0) {
      if (b + 1 < nb) {
        int jn = j0 + 32;
        tdm_load_tile_bf16(lds_off_u32(&Ks[cur ^ 1][0][0]),
                           Kbf + (size_t)jn * (NKV * HD) + kvh * HD,
                           32, HD, NKV * HD, NKV * HD, S_LEN, 5);
        tdm_load_tile_bf16(lds_off_u32(&Vs[cur ^ 1][0][0]),
                           Vt + (size_t)kvh * HD * S_LEN + jn,
                           HD, 32, S_LEN, S_LEN, HD, 3);
        __builtin_amdgcn_s_wait_tensorcnt(2);
      } else {
        __builtin_amdgcn_s_wait_tensorcnt(0);
      }
    }
    __syncthreads();

    if (j0 <= q0 + 15) {                 // wave-uniform guard
      v8f s0 = vzero, s1 = vzero;
#pragma unroll
      for (int c = 0; c < 4; ++c) {
        s0 = wmma_bf16(qf[c], load_b_frag(&Ks[cur][0][c * 32],  HD + 2, lane), s0);
        s1 = wmma_bf16(qf[c], load_b_frag(&Ks[cur][16][c * 32], HD + 2, lane), s1);
      }
#pragma unroll
      for (int r = 0; r < 8; ++r) {
        int row_l = r + half * 8;
        int row_g = q0 + row_l;
        float e0 = s0[r], e1 = s1[r];
        if (j0 + lm > row_g)      e0 = -1e30f;
        if (j0 + 16 + lm > row_g) e1 = -1e30f;
        float mx = fmaxf(e0, e1);
#pragma unroll
        for (int off = 8; off; off >>= 1) mx = fmaxf(mx, __shfl_xor(mx, off, 32));
        float mnew  = fmaxf(m_i[r], mx);
        float alpha = __expf(m_i[r] - mnew);
        float p0 = __expf(e0 - mnew);
        float p1 = __expf(e1 - mnew);
        float sum = p0 + p1;
#pragma unroll
        for (int off = 8; off; off >>= 1) sum += __shfl_xor(sum, off, 32);
        l_i[r] = l_i[r] * alpha + sum;
        m_i[r] = mnew;
#pragma unroll
        for (int nt = 0; nt < 8; ++nt) acc[nt][r] *= alpha;
        Ps[w][row_l][lm]      = (__bf16)p0;
        Ps[w][row_l][lm + 16] = (__bf16)p1;
      }
      v16bf pf = load_a_frag(&Ps[w][0][0], 32, lane);
#pragma unroll
      for (int nt = 0; nt < 8; ++nt)
        acc[nt] = wmma_bf16(pf, load_b_frag(&Vs[cur][nt * 16][0], 32 + 2, lane), acc[nt]);
    }
    __syncthreads();
  }

#pragma unroll
  for (int r = 0; r < 8; ++r) {
    float inv = 1.0f / l_i[r];
    int row_g = q0 + r + half * 8;
#pragma unroll
    for (int nt = 0; nt < 8; ++nt)
      ctx[(size_t)row_g * (NH * HD) + head * HD + nt * 16 + lm] =
          (__bf16)(acc[nt][r] * inv);
  }
}

// ---------------------------------------------------------------------------
extern "C" void kernel_launch(void* const* d_in, const int* in_sizes, int n_in,
                              void* d_out, int out_size, void* d_ws, size_t ws_size,
                              hipStream_t stream) {
  const float* X  = (const float*)d_in[0];   // [2048, 4096]
  const float* Wq = (const float*)d_in[1];   // [4096, 4096]
  const float* Wk = (const float*)d_in[2];   // [4096, 1024]
  const float* Wv = (const float*)d_in[3];   // [4096, 1024]
  const float* Wo = (const float*)d_in[4];   // [4096, 4096]
  // d_in[5] = additive causal mask (applied analytically in attn_kernel)

  char* ws = (char*)d_ws;
  const size_t MB = 1024 * 1024;
  float*  Qb    = (float*)(ws + 0);           // 32 MB fp32 [2048][4096]
  float*  Kb    = (float*)(ws + 32 * MB);     //  8 MB fp32 [2048][1024]
  float*  Vb    = (float*)(ws + 40 * MB);     //  8 MB fp32 [2048][1024]
  __bf16* Xbf   = (__bf16*)(ws + 48 * MB);    // 16 MB bf16 [2048][4096]
  __bf16* Wqt   = (__bf16*)(ws + 64 * MB);    // 32 MB bf16 [4096][4096] (transposed)
  __bf16* Wkt   = (__bf16*)(ws + 96 * MB);    //  8 MB bf16 [1024][4096]
  __bf16* Wvt   = (__bf16*)(ws + 104 * MB);   //  8 MB bf16 [1024][4096]
  __bf16* Wot   = (__bf16*)(ws + 112 * MB);   // 32 MB bf16 [4096][4096]
  __bf16* ctxbf = (__bf16*)(ws + 144 * MB);   // 16 MB bf16 [2048][4096]
  __bf16* Qbf   = (__bf16*)(ws + 160 * MB);   // 16 MB bf16 [2048][4096]
  __bf16* Kbf   = (__bf16*)(ws + 176 * MB);   //  4 MB bf16 [2048][1024]
  __bf16* Vtb   = (__bf16*)(ws + 180 * MB);   //  4 MB bf16 [8][128][2048]
  float*  out   = (float*)d_out;

  // 1) precision/layout prep
  {
    int n4 = (S_LEN * HID) / 4;
    f32_to_bf16_kernel<<<(n4 + 255) / 256, 256, 0, stream>>>(X, Xbf, n4);
    dim3 tb(32, 8);
    transpose_to_bf16_kernel<<<dim3(HID / 32, HID / 32), tb, 0, stream>>>(Wq, Wqt, HID, HID);
    transpose_to_bf16_kernel<<<dim3((NKV * HD) / 32, HID / 32), tb, 0, stream>>>(Wk, Wkt, HID, NKV * HD);
    transpose_to_bf16_kernel<<<dim3((NKV * HD) / 32, HID / 32), tb, 0, stream>>>(Wv, Wvt, HID, NKV * HD);
    transpose_to_bf16_kernel<<<dim3(HID / 32, HID / 32), tb, 0, stream>>>(Wo, Wot, HID, HID);
  }

  dim3 blk(128);
  // 2) QKV projections (TDM-staged bf16 WMMA GEMM, fp32 out)
  gemm_bf16_tdm<<<dim3((NH * HD) / GBN, S_LEN / GBM), blk, 0, stream>>>(
      Xbf, Wqt, Qb, S_LEN, NH * HD, HID);
  gemm_bf16_tdm<<<dim3((NKV * HD) / GBN, S_LEN / GBM), blk, 0, stream>>>(
      Xbf, Wkt, Kb, S_LEN, NKV * HD, HID);
  gemm_bf16_tdm<<<dim3((NKV * HD) / GBN, S_LEN / GBM), blk, 0, stream>>>(
      Xbf, Wvt, Vb, S_LEN, NKV * HD, HID);
  // 3) RoPE (fp32 math) -> bf16 Q (pre-scaled), bf16 K
  int rope_total = S_LEN * (NH + NKV) * (HD / 2);
  rope_bf16_kernel<<<(rope_total + 255) / 256, 256, 0, stream>>>(Qb, Kb, Qbf, Kbf);
  // 3b) V -> bf16, transposed per KV head
  v_transpose_kernel<<<dim3(S_LEN / 32, HD / 32, NKV), dim3(32, 8), 0, stream>>>(Vb, Vtb);
  // 4) Flash attention (TDM-fed) -> ctx (bf16)
  attn_kernel<<<dim3(S_LEN / 64, NH), blk, 0, stream>>>(Qbf, Kbf, Vtb, ctxbf);
  // 5) Output projection
  gemm_bf16_tdm<<<dim3(HID / GBN, S_LEN / GBM), blk, 0, stream>>>(
      ctxbf, Wot, out, S_LEN, HID, HID);
}
